// CapsuleNet_53626961658066
// MI455X (gfx1250) — matrix operand order repaired and checked by reading, not compiled, not run on hardware
//
#include <hip/hip_runtime.h>
#include <hip/hip_bf16.h>

// ---------------------------------------------------------------------------
// CapsuleNet forward for MI455X (gfx1250, wave32, WMMA).
// Dominant cost: conv1 as implicit GEMM M=9216, N=256, K=20736 (~98 GFLOP)
// -> f16 WMMA (v_wmma_f32_16x16x32_f16) with double-buffered async global->LDS
//    staging (GLOBAL_LOAD_ASYNC_TO_LDS_B128 / ASYNCcnt).
// ---------------------------------------------------------------------------

typedef __attribute__((ext_vector_type(16))) _Float16 v16h;
typedef __attribute__((ext_vector_type(8)))  float    v8f;

union FragU {
    unsigned int u[8];
    v16h h;
};

#define CONV1_K  20736   // 81 * 256
#define CONV1_KC 648     // K / 32

// ------------------------- conv0: 1ch 28x28 -> 256ch 20x20 (9x9, VALID) -----
// Output layout: act[b][(iy*20+ix)][oc] as f16 (NHWC) so conv1's implicit-GEMM
// A gather is contiguous over 32 input channels per K-chunk.
__global__ void conv0_kernel(const float* __restrict__ x,
                             const float* __restrict__ w,
                             const float* __restrict__ bias,
                             _Float16* __restrict__ act) {
    __shared__ float patch[81];
    const int blk = blockIdx.x;            // b*400 + pos
    const int b   = blk / 400;
    const int pos = blk % 400;
    const int iy  = pos / 20, ix = pos % 20;
    const int t   = threadIdx.x;
    if (t < 81) {
        const int ky = t / 9, kx = t % 9;
        patch[t] = x[b * 784 + (iy + ky) * 28 + (ix + kx)];
    }
    __syncthreads();
    const int oc = t;                       // 256 threads = 256 out channels
    float s = bias[oc];
    const float* wp = w + oc * 81;
#pragma unroll 9
    for (int k = 0; k < 81; ++k) s += patch[k] * wp[k];
    s = fmaxf(s, 0.0f);
    act[((size_t)blk << 8) + oc] = (_Float16)s;
}

// -------------- pack conv1 weights [OC][IC][9][9] f32 -> [OC][K] f16 --------
// k = (ky*9+kx)*256 + ic  (spatial-major, ic fastest -> contiguous K chunks)
__global__ void pack_conv1_w_kernel(const float* __restrict__ src,
                                    _Float16* __restrict__ dst) {
    const int i  = blockIdx.x * 256 + threadIdx.x;   // 0 .. 5308415
    const int oc = i / CONV1_K;
    const int r  = i % CONV1_K;
    const int s  = r >> 8;        // tap 0..80
    const int ic = r & 255;
    dst[i] = (_Float16)src[((size_t)(oc * 256 + ic)) * 81 + s];
}

// -------------- pack decoder weights [K][N] f32 -> col-major [N][K] f16 -----
__global__ void pack_colmajor_kernel(const float* __restrict__ src,
                                     _Float16* __restrict__ dst,
                                     int K, int N) {
    const int i = blockIdx.x * 256 + threadIdx.x;    // 0 .. N*K-1
    const int n = i / K;
    const int k = i % K;
    dst[i] = (_Float16)src[(size_t)k * N + n];
}

// ---------------- conv1 implicit GEMM via WMMA (dominant kernel) ------------
// C[m][n] = relu(sum_k A[m][k]*B[k][n] + bias[n]);  m=(b,p), n=oc
// Block tile 64(M) x 256(N); per-wave 16 x 128 (8 accumulators); K chunk = 32.
// Double-buffered LDS tiles staged with GLOBAL_LOAD_ASYNC_TO_LDS_B128 so the
// DMA for chunk k+1 overlaps the WMMAs of chunk k (tracked by ASYNCcnt).
__global__ void conv1_gemm_kernel(const _Float16* __restrict__ act,
                                  const _Float16* __restrict__ bpack,
                                  const float* __restrict__ bias,
                                  float* __restrict__ feat) {
    __shared__ __align__(16) _Float16 lA[2][64 * 40];    // stride 40 (16B pad)
    __shared__ __align__(16) _Float16 lB[2][256 * 40];   // stride 40 (16B pad)

    const int tid  = threadIdx.x;
    const int m0   = blockIdx.x * 64;
    const int lane = tid & 31;
    const int wv   = tid >> 5;
    const int wm   = wv >> 1;      // 0..3  (M sub-tile)
    const int wn   = wv & 1;       // 0..1  (N half: 128 cols)
    const int frow  = lane & 15;
    const int khalf = lane >> 4;

    // A staging: each thread copies one 16B run (8 f16)
    const int arow = tid >> 2;            // 0..63
    const int akq  = (tid & 3) << 3;      // 0,8,16,24
    const int m = m0 + arow;
    const int b = m / 36, p = m % 36;
    const int oy = p / 6, ox = p % 6;
    const size_t abase = ((size_t)b * 400) << 8;

    // B staging: thread tid handles column n = tid (32 f16 contiguous, 4x16B)
    const _Float16* bsrc0 = bpack + (size_t)tid * CONV1_K;

    // LDS byte addresses (flat shared pointer truncates to LDS offset)
    const unsigned aoff[2] = {
        (unsigned)(unsigned long long)&lA[0][arow * 40 + akq],
        (unsigned)(unsigned long long)&lA[1][arow * 40 + akq]};
    const unsigned boff[2] = {
        (unsigned)(unsigned long long)&lB[0][tid * 40],
        (unsigned)(unsigned long long)&lB[1][tid * 40]};

    v8f z = {};
    v8f acc[8] = {z, z, z, z, z, z, z, z};

    auto stage = [&](int kc, int sel) {
        const int s   = kc >> 3;           // tap 0..80
        const int ic0 = (kc & 7) << 5;     // 0..224
        const int ky = s / 9, kx = s % 9;
        const int iy = 2 * oy + ky, ix = 2 * ox + kx;
        const _Float16* asrc =
            act + abase + (((size_t)(iy * 20 + ix)) << 8) + ic0 + akq;
        asm volatile("global_load_async_to_lds_b128 %0, %1, off"
                     :: "v"(aoff[sel]), "v"((unsigned long long)asrc)
                     : "memory");
        const _Float16* bsrc = bsrc0 + kc * 32;
#pragma unroll
        for (int q = 0; q < 4; ++q) {
            asm volatile("global_load_async_to_lds_b128 %0, %1, off"
                         :: "v"(boff[sel] + q * 16),
                            "v"((unsigned long long)(bsrc + q * 8))
                         : "memory");
        }
    };

    stage(0, 0);
    asm volatile("s_wait_asynccnt 0x0" ::: "memory");
    __syncthreads();

    for (int kc = 0; kc < CONV1_KC; ++kc) {
        const int sel = kc & 1;
        if (kc + 1 < CONV1_KC) stage(kc + 1, sel ^ 1);   // overlap with WMMA

        // A fragment (ISA: vgpr j -> kp = (j>>2)*16 + khalf*8 + (j&3)*2)
        FragU a;
#pragma unroll
        for (int j = 0; j < 8; ++j) {
            const int kp = ((j >> 2) << 4) + (khalf << 3) + ((j & 3) << 1);
            a.u[j] = *(const unsigned int*)&lA[sel][(wm * 16 + frow) * 40 + kp];
        }
#pragma unroll
        for (int i = 0; i < 8; ++i) {
            FragU bf;
            const int n = wn * 128 + i * 16 + frow;
#pragma unroll
            for (int j = 0; j < 8; ++j) {
                const int kp = ((j >> 2) << 4) + (khalf << 3) + ((j & 3) << 1);  // B: same k order as A
                bf.u[j] = *(const unsigned int*)&lB[sel][n * 40 + kp];
            }
            acc[i] = __builtin_amdgcn_wmma_f32_16x16x32_f16(
                false, a.h, false, bf.h, (short)0, acc[i], false, false);
        }
        asm volatile("s_wait_asynccnt 0x0" ::: "memory");
        __syncthreads();
    }

    // epilogue: bias + relu, write feat[m][n] (f32, row-major [9216][256])
    const int mrow0 = m0 + wm * 16 + (khalf << 3);
#pragma unroll
    for (int i = 0; i < 8; ++i) {
        const int n = wn * 128 + i * 16 + frow;
        const float bz = bias[n];
#pragma unroll
        for (int r = 0; r < 8; ++r) {
            feat[(size_t)(mrow0 + r) * 256 + n] = fmaxf(acc[i][r] + bz, 0.0f);
        }
    }
}

// -------- generic 256-row WMMA GEMM: C = act(A[256][K] * Bcol[N][K] + bias) -
// ACT: 0 = relu, 1 = sigmoid.  OUT16: write f16 (for chained layers) or f32.
template <int ACT, bool OUT16>
__global__ void gemm16_kernel(const _Float16* __restrict__ A,
                              const _Float16* __restrict__ Bc,
                              const float* __restrict__ bias,
                              _Float16* __restrict__ outh,
                              float* __restrict__ outf,
                              int N, int K) {
    __shared__ __align__(16) _Float16 lA[32 * 36];
    __shared__ __align__(16) _Float16 lB[256 * 40];

    const int tid  = threadIdx.x;
    const int m0   = blockIdx.x * 32;
    const int n0   = blockIdx.y * 256;
    const int lane = tid & 31;
    const int wv   = tid >> 5;
    const int wm   = wv >> 2, wn = wv & 3;
    const int frow  = lane & 15;
    const int khalf = lane >> 4;

    const int arow = tid >> 3;
    const int akq  = (tid & 7) << 2;
    const int ncol = n0 + tid;
    const bool bvalid = (ncol < N);
    const int chunks = K >> 5;

    v8f z = {};
    v8f acc[4] = {z, z, z, z};

    for (int kc = 0; kc < chunks; ++kc) {
        *(uint2*)&lA[arow * 36 + akq] =
            *(const uint2*)(A + (size_t)(m0 + arow) * K + kc * 32 + akq);
        if (bvalid) {
            const _Float16* bsrc = Bc + (size_t)ncol * K + kc * 32;
#pragma unroll
            for (int q = 0; q < 4; ++q)
                *(uint4*)&lB[tid * 40 + q * 8] = *(const uint4*)(bsrc + q * 8);
            __builtin_prefetch(bsrc + 32, 0, 1);
        }
        __syncthreads();

        FragU a;
#pragma unroll
        for (int j = 0; j < 8; ++j) {
            const int kp = ((j >> 2) << 4) + (khalf << 3) + ((j & 3) << 1);
            a.u[j] = *(const unsigned int*)&lA[(wm * 16 + frow) * 36 + kp];
        }
#pragma unroll
        for (int i = 0; i < 4; ++i) {
            const int nb = n0 + wn * 64 + i * 16;
            if (nb + 16 <= N) {
                FragU bf;
                const int nl = wn * 64 + i * 16 + frow;
#pragma unroll
                for (int j = 0; j < 8; ++j) {
                    const int kp = ((j >> 2) << 4) + (khalf << 3) + ((j & 3) << 1);
                    bf.u[j] = *(const unsigned int*)&lB[nl * 40 + kp];
                }
                acc[i] = __builtin_amdgcn_wmma_f32_16x16x32_f16(
                    false, a.h, false, bf.h, (short)0, acc[i], false, false);
            }
        }
        __syncthreads();
    }

    const int mrow0 = m0 + wm * 16 + (khalf << 3);
#pragma unroll
    for (int i = 0; i < 4; ++i) {
        const int nb = n0 + wn * 64 + i * 16;
        if (nb + 16 <= N) {
            const int n = nb + frow;
            const float bz = bias[n];
#pragma unroll
            for (int r = 0; r < 8; ++r) {
                float v = acc[i][r] + bz;
                if (ACT == 0) v = fmaxf(v, 0.0f);
                else          v = 1.0f / (1.0f + __expf(-v));
                if (OUT16) outh[(size_t)(mrow0 + r) * N + n] = (_Float16)v;
                else       outf[(size_t)(mrow0 + r) * N + n] = v;
            }
        }
    }
}

// ------------------- CFC: per-patch Linear(256->8) + squash ----------------
__global__ void cfc_kernel(const float* __restrict__ feat,
                           const float* __restrict__ w,
                           const float* __restrict__ bias,
                           float* __restrict__ u) {
    const int idx = blockIdx.x * 256 + threadIdx.x;   // 0..9215 = b*36+p
    const int p = idx % 36;
    float s[8];
#pragma unroll
    for (int d = 0; d < 8; ++d) s[d] = bias[p * 8 + d];
    const float* f  = feat + (size_t)idx * 256;
    const float* wp = w + (size_t)p * 256 * 8;
    for (int c = 0; c < 256; ++c) {
        const float fv = f[c];
#pragma unroll
        for (int d = 0; d < 8; ++d) s[d] += fv * wp[c * 8 + d];
    }
    float sq = 0.0f;
#pragma unroll
    for (int d = 0; d < 8; ++d) sq += s[d] * s[d];
    const float scale = (sq / (1.0f + sq)) * rsqrtf(sq + 1e-12f);
#pragma unroll
    for (int d = 0; d < 8; ++d) u[(size_t)idx * 8 + d] = s[d] * scale;
}

// ------------------- priors[c][b][p][o] = sum_d u[b][p][d]*W[c][p][d][o] ----
__global__ void priors_kernel(const float* __restrict__ u,
                              const float* __restrict__ rw,
                              float* __restrict__ priors) {
    const int idx = blockIdx.x * 256 + threadIdx.x;   // 0..92159 = (c*256+b)*36+p
    const int c = idx / (256 * 36);
    const int r = idx % (256 * 36);
    const int b = r / 36, p = r % 36;
    float ud[8];
#pragma unroll
    for (int d = 0; d < 8; ++d) ud[d] = u[((size_t)(b * 36 + p)) * 8 + d];
    const float* w = rw + ((size_t)(c * 36 + p)) * 8 * 16;
    float* out = priors + (size_t)idx * 16;
#pragma unroll
    for (int o = 0; o < 16; ++o) {
        float s = 0.0f;
#pragma unroll
        for (int d = 0; d < 8; ++d) s += ud[d] * w[d * 16 + o];
        out[o] = s;
    }
}

// ------------------- dynamic routing (3 iters), one thread per (c,b) --------
__global__ void routing_kernel(const float* __restrict__ priors,
                               float* __restrict__ vout) {
    const int idx = blockIdx.x * 256 + threadIdx.x;   // 0..2559 = c*256+b
    const float* pr = priors + (size_t)idx * 36 * 16;
    float logit[36];
    for (int p = 0; p < 36; ++p) logit[p] = 0.0f;
    float v[16];
    for (int it = 0; it < 3; ++it) {
        float mx = logit[0];
        for (int p = 1; p < 36; ++p) mx = fmaxf(mx, logit[p]);
        float prob[36];
        float se = 0.0f;
        for (int p = 0; p < 36; ++p) { prob[p] = __expf(logit[p] - mx); se += prob[p]; }
        const float inv = 1.0f / se;
        float s[16];
#pragma unroll
        for (int o = 0; o < 16; ++o) s[o] = 0.0f;
        for (int p = 0; p < 36; ++p) {
            const float w = prob[p] * inv;
            const float* pp = pr + p * 16;
#pragma unroll
            for (int o = 0; o < 16; ++o) s[o] += w * pp[o];
        }
        float sq = 0.0f;
#pragma unroll
        for (int o = 0; o < 16; ++o) sq += s[o] * s[o];
        const float scale = (sq / (1.0f + sq)) * rsqrtf(sq + 1e-12f);
#pragma unroll
        for (int o = 0; o < 16; ++o) v[o] = s[o] * scale;
        if (it < 2) {
            for (int p = 0; p < 36; ++p) {
                const float* pp = pr + p * 16;
                float d = 0.0f;
#pragma unroll
                for (int o = 0; o < 16; ++o) d += pp[o] * v[o];
                logit[p] += d;
            }
        }
    }
#pragma unroll
    for (int o = 0; o < 16; ++o) vout[(size_t)idx * 16 + o] = v[o];
}

// -------- classes softmax + argmax one-hot mask -> masked f16 [B][160] ------
__global__ void classes_kernel(const float* __restrict__ vout,
                               float* __restrict__ out_classes,
                               _Float16* __restrict__ maskedh) {
    const int b = threadIdx.x;   // 256 threads, one block
    float norm[10];
    for (int c = 0; c < 10; ++c) {
        const float* v = vout + ((size_t)(c * 256 + b)) * 16;
        float sq = 0.0f;
#pragma unroll
        for (int o = 0; o < 16; ++o) sq += v[o] * v[o];
        norm[c] = sqrtf(sq);
    }
    float mx = norm[0];
    int amax = 0;
    for (int c = 1; c < 10; ++c)
        if (norm[c] > mx) { mx = norm[c]; amax = c; }
    float e[10], se = 0.0f;
    for (int c = 0; c < 10; ++c) { e[c] = __expf(norm[c] - mx); se += e[c]; }
    const float inv = 1.0f / se;
    for (int c = 0; c < 10; ++c) out_classes[b * 10 + c] = e[c] * inv;
    for (int c = 0; c < 10; ++c) {
        const float* v = vout + ((size_t)(c * 256 + b)) * 16;
#pragma unroll
        for (int o = 0; o < 16; ++o)
            maskedh[(size_t)b * 160 + c * 16 + o] =
                (c == amax) ? (_Float16)v[o] : (_Float16)0.0f;
    }
}

// ---------------------------------------------------------------------------
extern "C" void kernel_launch(void* const* d_in, const int* in_sizes, int n_in,
                              void* d_out, int out_size, void* d_ws, size_t ws_size,
                              hipStream_t stream) {
    const float* x       = (const float*)d_in[0];
    const float* conv0_w = (const float*)d_in[1];
    const float* conv0_b = (const float*)d_in[2];
    const float* conv1_w = (const float*)d_in[3];
    const float* conv1_b = (const float*)d_in[4];
    const float* cfc_w   = (const float*)d_in[5];
    const float* cfc_b   = (const float*)d_in[6];
    const float* route_w = (const float*)d_in[7];
    const float* dec_w1  = (const float*)d_in[8];
    const float* dec_b1  = (const float*)d_in[9];
    const float* dec_w2  = (const float*)d_in[10];
    const float* dec_b2  = (const float*)d_in[11];
    const float* dec_w3  = (const float*)d_in[12];
    const float* dec_b3  = (const float*)d_in[13];

    float* out_classes = (float*)d_out;           // [256][10]
    float* out_recon   = (float*)d_out + 2560;    // [256][784]

    char* ws = (char*)d_ws;
    size_t off = 0;
    auto carve = [&](size_t bytes) -> char* {
        char* p = ws + off;
        off += (bytes + 255) & ~(size_t)255;
        return p;
    };
    _Float16* act_h   = (_Float16*)carve((size_t)256 * 400 * 256 * 2);  // 52.4 MB
    _Float16* bpack   = (_Float16*)carve((size_t)256 * CONV1_K * 2);    // 10.6 MB
    float*    feat    = (float*)   carve((size_t)9216 * 256 * 4);       // 9.4 MB
    float*    u       = (float*)   carve((size_t)9216 * 8 * 4);
    float*    priors  = (float*)   carve((size_t)10 * 256 * 36 * 16 * 4);
    float*    vout    = (float*)   carve((size_t)10 * 256 * 16 * 4);
    _Float16* maskedh = (_Float16*)carve((size_t)256 * 160 * 2);
    _Float16* w1p     = (_Float16*)carve((size_t)512 * 160 * 2);
    _Float16* w2p     = (_Float16*)carve((size_t)1024 * 512 * 2);
    _Float16* w3p     = (_Float16*)carve((size_t)784 * 1024 * 2);
    _Float16* h1      = (_Float16*)carve((size_t)256 * 512 * 2);
    _Float16* h2      = (_Float16*)carve((size_t)256 * 1024 * 2);
    (void)ws_size; (void)in_sizes; (void)n_in; (void)out_size;

    // 1) conv0 -> f16 NHWC activations
    conv0_kernel<<<256 * 400, 256, 0, stream>>>(x, conv0_w, conv0_b, act_h);

    // 2) weight packs (f32 -> f16, GEMM-friendly layouts)
    pack_conv1_w_kernel<<<(256 * CONV1_K) / 256, 256, 0, stream>>>(conv1_w, bpack);
    pack_colmajor_kernel<<<(512 * 160) / 256, 256, 0, stream>>>(dec_w1, w1p, 160, 512);
    pack_colmajor_kernel<<<(1024 * 512) / 256, 256, 0, stream>>>(dec_w2, w2p, 512, 1024);
    pack_colmajor_kernel<<<(784 * 1024) / 256, 256, 0, stream>>>(dec_w3, w3p, 1024, 784);

    // 3) conv1 implicit GEMM (WMMA + async LDS staging) -> feat[b][p][c]
    conv1_gemm_kernel<<<9216 / 64, 256, 0, stream>>>(act_h, bpack, conv1_b, feat);

    // 4) capsule head
    cfc_kernel<<<9216 / 256, 256, 0, stream>>>(feat, cfc_w, cfc_b, u);
    priors_kernel<<<(10 * 256 * 36) / 256, 256, 0, stream>>>(u, route_w, priors);
    routing_kernel<<<(10 * 256) / 256, 256, 0, stream>>>(priors, vout);
    classes_kernel<<<1, 256, 0, stream>>>(vout, out_classes, maskedh);

    // 5) decoder: 160->512 relu, 512->1024 relu, 1024->784 sigmoid (WMMA)
    gemm16_kernel<0, true><<<dim3(8, 2), 256, 0, stream>>>(
        maskedh, w1p, dec_b1, h1, nullptr, 512, 160);
    gemm16_kernel<0, true><<<dim3(8, 4), 256, 0, stream>>>(
        h1, w2p, dec_b2, h2, nullptr, 1024, 512);
    gemm16_kernel<1, false><<<dim3(8, 4), 256, 0, stream>>>(
        h2, w3p, dec_b3, nullptr, out_recon, 784, 1024);
}
